// RetinaNetDecoder_86397562126382
// MI455X (gfx1250) — compile-verified, matrix operand order checked by reading, not run on hardware
//
#include <hip/hip_runtime.h>
#include <hip/hip_bf16.h>
#include <stdint.h>

#define BATCH 2
#define NANCH 8192
#define NCLS  80
#define KNMS  2048          // top-K window for NMS (>> 100 survivors needed)
#define KCHUNK (KNMS / 16)  // uint16 chunks per mask row
#define MAXDET 100
#define IMW 1024
#define IMH 1024
#define SCORE_THR 0.05f
#define NMS_THR 0.5f

typedef __attribute__((ext_vector_type(16))) _Float16 v16h;
typedef __attribute__((ext_vector_type(8)))  float    v8f;

typedef __attribute__((address_space(1))) int as1_int;
typedef __attribute__((address_space(3))) int as3_int;

#if defined(__has_builtin)
#if __has_builtin(__builtin_amdgcn_global_load_async_to_lds_b32) && \
    __has_builtin(__builtin_amdgcn_s_wait_asynccnt)
#define USE_ASYNC_LDS 1
#endif
#if __has_builtin(__builtin_amdgcn_ballot_w32)
#define BALLOT32(p) __builtin_amdgcn_ballot_w32((p))
#endif
#endif
#ifndef USE_ASYNC_LDS
#define USE_ASYNC_LDS 0
#endif
#ifndef BALLOT32
#define BALLOT32(p) ((unsigned)__ballot((p)))
#endif

// ---------------------------------------------------------------------------
// Kernel 1: per-anchor class max/argmax + box decode.  Pure HBM-bandwidth.
// ---------------------------------------------------------------------------
__global__ __launch_bounds__(256) void decode_kernel(
    const float* __restrict__ cls, const float* __restrict__ reg,
    const float* __restrict__ anc, float* __restrict__ scoreWs,
    int* __restrict__ clsWs, int4* __restrict__ boxWs)
{
    int t = blockIdx.x * 256 + threadIdx.x;
    if (t >= BATCH * NANCH) return;

    const float4* r4 = (const float4*)(cls + (size_t)t * NCLS);
    float best = -1.0f;
    int   bi   = 0;
#pragma unroll
    for (int q = 0; q < NCLS / 4; ++q) {
        float4 v = r4[q];
        if (v.x > best) { best = v.x; bi = 4 * q + 0; }
        if (v.y > best) { best = v.y; bi = 4 * q + 1; }
        if (v.z > best) { best = v.z; bi = 4 * q + 2; }
        if (v.w > best) { best = v.w; bi = 4 * q + 3; }
    }

    float4 rg = ((const float4*)reg)[t];
    float4 an = ((const float4*)anc)[t];
    float aw  = an.z - an.x, ah = an.w - an.y;
    float acx = an.x + 0.5f * aw, acy = an.y + 0.5f * ah;
    float dx = rg.x * 0.1f, dy = rg.y * 0.1f;
    float dw = rg.z * 0.2f, dh = rg.w * 0.2f;
    float pw  = expf(dw) * aw, ph = expf(dh) * ah;
    float pcx = dx * aw + acx, pcy = dy * ah + acy;

    int x1 = (int)(pcx - 0.5f * pw);   // trunc-toward-zero == astype(int32)
    int y1 = (int)(pcy - 0.5f * ph);
    int x2 = (int)(pcx + 0.5f * pw);
    int y2 = (int)(pcy + 0.5f * ph);
    x1 = max(x1, 0);       y1 = max(y1, 0);
    x2 = min(x2, IMW - 1); y2 = min(y2, IMH - 1);

    scoreWs[t] = best;
    clsWs[t]   = bi;
    boxWs[t]   = make_int4(x1, y1, x2, y2);
}

// ---------------------------------------------------------------------------
// Kernel 2: full 8192-key bitonic sort in 64KB LDS, one workgroup per batch.
// key = (~score_bits)<<32 | index  -> ascending sort == stable argsort(-score).
// Score staging goes through the gfx1250 async global->LDS path (ASYNCcnt).
// ---------------------------------------------------------------------------
__global__ __launch_bounds__(1024) void sort_kernel(
    const float* __restrict__ scoreWs, unsigned* __restrict__ sortIdx)
{
    __shared__ unsigned long long keys[NANCH];   // 64 KB
    const int b   = blockIdx.x;
    const int tid = threadIdx.x;
    const float* sc = scoreWs + (size_t)b * NANCH;

    // staging buffer aliases the upper half of the key array (u32 slots 8192..)
    float* staging = ((float*)keys) + NANCH;
    float  sreg[8];
#if USE_ASYNC_LDS
#pragma unroll
    for (int r = 0; r < 8; ++r) {
        int i = tid + r * 1024;
        __builtin_amdgcn_global_load_async_to_lds_b32(
            (as1_int*)(sc + i),
            (as3_int*)(staging + i),
            0, 0);
    }
    __builtin_amdgcn_s_wait_asynccnt(0);
#pragma unroll
    for (int r = 0; r < 8; ++r) sreg[r] = staging[tid + r * 1024];
#else
#pragma unroll
    for (int r = 0; r < 8; ++r) sreg[r] = sc[tid + r * 1024];
#endif
    __syncthreads();   // staging fully consumed before keys overwrite it
#pragma unroll
    for (int r = 0; r < 8; ++r) {
        int i = tid + r * 1024;
        unsigned sb = __float_as_uint(sreg[r]);
        keys[i] = ((unsigned long long)(~sb) << 32) | (unsigned)i;
    }
    __syncthreads();

    for (unsigned k = 2; k <= NANCH; k <<= 1) {
        for (unsigned j = k >> 1; j > 0; j >>= 1) {
#pragma unroll
            for (int s = 0; s < 4; ++s) {
                unsigned t = (unsigned)tid + (unsigned)s * 1024u;   // t < 4096
                unsigned i = (t / j) * (2u * j) + (t % j);
                unsigned l = i + j;
                unsigned long long a = keys[i], c = keys[l];
                bool up = ((i & k) == 0);
                if ((a > c) == up) { keys[i] = c; keys[l] = a; }
            }
            __syncthreads();
        }
    }
#pragma unroll
    for (int r = 0; r < 8; ++r) {
        int i = tid + r * 1024;
        sortIdx[(size_t)b * NANCH + i] = (unsigned)(keys[i] & 0xFFFFFFFFu);
    }
}

// ---------------------------------------------------------------------------
// Kernel 3: 16x16 suppression-bit tiles over the top-KNMS sorted candidates.
// union = (area_i + area_j) - inter; the outer-sum term is a rank-2 matmul on
// the matrix engine (v_wmma_f32_16x16x32_f16), areas scaled by 2^-10 to stay
// in f16 range.  D layout: vgpr k -> rows {k, k+8}, lanes -> cols 0..15.
// Division-free threshold: iou>0.5 <=> uni>0 && 2*inter>uni.
// One wave per tile; sup bits compressed with ballot into uint16 row chunks.
// ---------------------------------------------------------------------------
__global__ __launch_bounds__(32) void nms_mask_kernel(
    const int* __restrict__ clsWs, const int4* __restrict__ boxWs,
    const unsigned* __restrict__ sortIdx, unsigned short* __restrict__ mask)
{
    const int bid   = blockIdx.x;
    const int tileJ = bid % KCHUNK;
    const int tileI = (bid / KCHUNK) % KCHUNK;
    const int b     = bid / (KCHUNK * KCHUNK);
    const int lane  = threadIdx.x;
    const int i0 = tileI * 16, j0 = tileJ * 16;
    unsigned short* mbase = mask + (size_t)b * KNMS * KCHUNK;

    if (tileJ < tileI) {            // j > i impossible: row chunk is all-zero
        if (lane < 16) mbase[(size_t)(i0 + lane) * KCHUNK + tileJ] = 0;
        return;
    }

    __shared__ float rx1[16], ry1[16], rx2[16], ry2[16], rar[16];
    __shared__ float qx1[16], qy1[16], qx2[16], qy2[16], qar[16];
    __shared__ int   rcl[16], qcl[16];
    {
        int p = (lane < 16) ? (i0 + lane) : (j0 + lane - 16);
        unsigned orig = sortIdx[(size_t)b * NANCH + p];
        int4 bx = boxWs[(size_t)b * NANCH + orig];
        int  cl = clsWs[(size_t)b * NANCH + orig];
        float fx1 = (float)bx.x, fy1 = (float)bx.y;
        float fx2 = (float)bx.z, fy2 = (float)bx.w;
        float ar = fmaxf(fx2 - fx1, 0.f) * fmaxf(fy2 - fy1, 0.f);
        if (lane < 16) {
            rx1[lane] = fx1; ry1[lane] = fy1; rx2[lane] = fx2; ry2[lane] = fy2;
            rar[lane] = ar;  rcl[lane] = cl;
        } else {
            int c = lane - 16;
            qx1[c] = fx1; qy1[c] = fy1; qx2[c] = fx2; qy2[c] = fy2;
            qar[c] = ar;  qcl[c] = cl;
        }
    }
    __syncthreads();

    // outer-sum (area_i + area_j) * 2^-10 on the matrix core
    const float SCI = 1.0f / 1024.0f;
    v16h a, bb;
    for (int e = 0; e < 16; ++e) { a[e] = (_Float16)0.0f; bb[e] = (_Float16)0.0f; }
    if (lane < 16) {
        a[0]  = (_Float16)(rar[lane] * SCI);  // A[m, K=0] = area_row*2^-10
        a[1]  = (_Float16)1.0f;               // A[m, K=1] = 1
        bb[0] = (_Float16)1.0f;               // B[K=0, n] = 1
        bb[1] = (_Float16)(qar[lane] * SCI);  // B[K=1, n] = area_col*2^-10
    }
    v8f acc;
    for (int e = 0; e < 8; ++e) acc[e] = 0.0f;
    acc = __builtin_amdgcn_wmma_f32_16x16x32_f16(
        false, a, false, bb, (short)0, acc, false, false);

    const int col   = lane & 15;
    const int rbase = (lane >> 4) * 8;
    const float c_x1 = qx1[col], c_y1 = qy1[col];
    const float c_x2 = qx2[col], c_y2 = qy2[col];
    const int   c_c  = qcl[col];

    unsigned bal[8];
#pragma unroll
    for (int kk = 0; kk < 8; ++kk) {
        int row = rbase + kk;
        float ix1 = fmaxf(rx1[row], c_x1);
        float iy1 = fmaxf(ry1[row], c_y1);
        float ix2 = fminf(rx2[row], c_x2);
        float iy2 = fminf(ry2[row], c_y2);
        float inter = fmaxf(ix2 - ix1, 0.f) * fmaxf(iy2 - iy1, 0.f);
        float uni   = acc[kk] * 1024.0f - inter;
        // iou > 0.5  <=>  uni > 0  &&  2*inter > uni   (no division)
        bool sup = (uni > 0.f) && (inter + inter > uni) &&
                   (rcl[row] == c_c) && ((j0 + col) > (i0 + row));
        bal[kk] = BALLOT32(sup);   // bits 0..15: row kk; bits 16..31: row kk+8
    }
    if (lane == 0) {
#pragma unroll
        for (int kk = 0; kk < 8; ++kk) {
            mbase[(size_t)(i0 + kk)     * KCHUNK + tileJ] = (unsigned short)(bal[kk] & 0xFFFFu);
            mbase[(size_t)(i0 + kk + 8) * KCHUNK + tileJ] = (unsigned short)(bal[kk] >> 16);
        }
    }
}

// ---------------------------------------------------------------------------
// Kernel 4: greedy scan over the bitmask, one wave per batch; emit top-100.
// ---------------------------------------------------------------------------
__global__ __launch_bounds__(32) void nms_greedy_kernel(
    const float* __restrict__ scoreWs, const int* __restrict__ clsWs,
    const int4* __restrict__ boxWs, const unsigned* __restrict__ sortIdx,
    const unsigned short* __restrict__ mask, float* __restrict__ out)
{
    const int b    = blockIdx.x;
    const int lane = threadIdx.x;
    __shared__ unsigned keepW[KNMS / 32];
    __shared__ int      accList[MAXDET];

    for (int w = lane; w < KNMS / 32; w += 32) {
        unsigned bits = 0;
        for (int t = 0; t < 32; ++t) {
            int j = w * 32 + t;
            unsigned orig = sortIdx[(size_t)b * NANCH + j];
            float s = scoreWs[(size_t)b * NANCH + orig];
            if (s > SCORE_THR) bits |= (1u << t);
        }
        keepW[w] = bits;
    }
    __syncthreads();

    int accepted = 0;
    for (int i = 0; i < KNMS && accepted < MAXDET; ++i) {
        unsigned kw = keepW[i >> 5];                 // uniform across the wave
        if (!((kw >> (i & 31)) & 1u)) continue;
        if (lane == 0) accList[accepted] = i;
        accepted++;
        const unsigned short* srow = mask + ((size_t)b * KNMS + i) * KCHUNK;
        for (int w = lane; w < KNMS / 32; w += 32) {
            unsigned sup = (unsigned)srow[2 * w] | ((unsigned)srow[2 * w + 1] << 16);
            keepW[w] &= ~sup;
        }
        __syncthreads();
    }
    __syncthreads();

    float* outS = out;                          // [B][100] scores
    float* outC = out + BATCH * MAXDET;         // [B][100] classes (as float)
    float* outB = out + 2 * BATCH * MAXDET;     // [B][100][4] boxes (as float)
    for (int d = lane; d < MAXDET; d += 32) {
        float s = -1.f, c = -1.f, b0 = -1.f, b1 = -1.f, b2 = -1.f, b3 = -1.f;
        if (d < accepted) {
            int i = accList[d];
            unsigned orig = sortIdx[(size_t)b * NANCH + i];
            s = scoreWs[(size_t)b * NANCH + orig];
            c = (float)clsWs[(size_t)b * NANCH + orig];
            int4 bx = boxWs[(size_t)b * NANCH + orig];
            b0 = (float)bx.x; b1 = (float)bx.y; b2 = (float)bx.z; b3 = (float)bx.w;
        }
        outS[b * MAXDET + d] = s;
        outC[b * MAXDET + d] = c;
        outB[(b * MAXDET + d) * 4 + 0] = b0;
        outB[(b * MAXDET + d) * 4 + 1] = b1;
        outB[(b * MAXDET + d) * 4 + 2] = b2;
        outB[(b * MAXDET + d) * 4 + 3] = b3;
    }
}

// ---------------------------------------------------------------------------
extern "C" void kernel_launch(void* const* d_in, const int* in_sizes, int n_in,
                              void* d_out, int out_size, void* d_ws, size_t ws_size,
                              hipStream_t stream)
{
    const float* cls = (const float*)d_in[0];   // [B, N, 80]
    const float* reg = (const float*)d_in[1];   // [B, N, 4]
    const float* anc = (const float*)d_in[2];   // [B, N, 4]

    char* p = (char*)d_ws;
    float* scoreWs = (float*)p;          p += (size_t)BATCH * NANCH * 4;
    int*   clsWs   = (int*)p;            p += (size_t)BATCH * NANCH * 4;
    int4*  boxWs   = (int4*)p;           p += (size_t)BATCH * NANCH * 16;
    unsigned* sortIdx = (unsigned*)p;    p += (size_t)BATCH * NANCH * 4;
    unsigned short* mask = (unsigned short*)p;   // B * 2048 * 128 u16 = 1 MB

    float* out = (float*)d_out;

    decode_kernel<<<(BATCH * NANCH) / 256, 256, 0, stream>>>(
        cls, reg, anc, scoreWs, clsWs, boxWs);
    sort_kernel<<<BATCH, 1024, 0, stream>>>(scoreWs, sortIdx);
    nms_mask_kernel<<<BATCH * KCHUNK * KCHUNK, 32, 0, stream>>>(
        clsWs, boxWs, sortIdx, mask);
    nms_greedy_kernel<<<BATCH, 32, 0, stream>>>(
        scoreWs, clsWs, boxWs, sortIdx, mask, out);
}
